// GraphEmbedding_4226247819265
// MI455X (gfx1250) — matrix operand order repaired and checked by reading, not compiled, not run on hardware
//
#include <hip/hip_runtime.h>

// ---------------- constants (match reference) ----------------
constexpr int NN = 100000;   // atoms
constexpr int EE = 400000;   // directed edges
constexpr int FF = 11;       // raw feature dim
constexpr int DV = 512;      // vector dim
constexpr int LL = 5;        // conv layers
constexpr int GG = 2048;     // graphs
constexpr int BM = 128, BN = 64, BK = 32;
constexpr int NPAD = ((NN + BM - 1) / BM) * BM;   // 100096, zero-padded rows in hb

// ---------------- types ----------------
typedef __attribute__((ext_vector_type(16))) __bf16 v16bf;
typedef __attribute__((ext_vector_type(8)))  float  v8f;

struct alignas(16) F4 { float x, y, z, w; };
struct alignas(16) U4 { unsigned a, b, c, d; };

union FragBF { F4 f4[2]; v16bf v; __device__ FragBF() {} };

static __device__ __forceinline__ unsigned f2bf(float f) {
    unsigned u = __float_as_uint(f);
    u += 0x7FFFu + ((u >> 16) & 1u);   // round-to-nearest-even
    return u >> 16;
}
static __device__ __forceinline__ unsigned pk2(float lo, float hi) {
    return f2bf(lo) | (f2bf(hi) << 16);
}
static __device__ __forceinline__ void atomAddF(float* p, float v) {
    unsafeAtomicAdd(p, v);             // global_atomic_add_f32 on gfx1250
}
// async 16B global -> LDS copy (per lane), tracked by ASYNCcnt
static __device__ __forceinline__ void asyncCp16(unsigned ldsAddr, const void* gAddr) {
    asm volatile("global_load_async_to_lds_b128 %0, %1, off"
                 :: "v"(ldsAddr), "v"(gAddr) : "memory");
}
static __device__ __forceinline__ void waitAsync0() {
    asm volatile("s_wait_asynccnt 0" ::: "memory");
}

// ---------------- small utility kernels ----------------
__global__ void zero_f32(float* __restrict__ p, int n) {
    int t = blockIdx.x * blockDim.x + threadIdx.x;
    if (t < n) p[t] = 0.0f;
}
__global__ void zero_i32(int* __restrict__ p, int n) {
    int t = blockIdx.x * blockDim.x + threadIdx.x;
    if (t < n) p[t] = 0;
}

__global__ void count_deg(const int* __restrict__ ei, int* __restrict__ degi) {
    int t = blockIdx.x * blockDim.x + threadIdx.x;
    if (t < EE) atomicAdd(&degi[ei[EE + t]], 1);
}

__global__ void norms_k(const int* __restrict__ degi, float* __restrict__ dinv,
                        float* __restrict__ selfn) {
    int t = blockIdx.x * blockDim.x + threadIdx.x;
    if (t < NN) {
        float dg = (float)degi[t] + 2.0f;
        float di = rsqrtf(dg);
        dinv[t]  = di;
        selfn[t] = 2.0f * di * di;
    }
}

__global__ void enorm_k(const int* __restrict__ ei, const float* __restrict__ dinv,
                        float* __restrict__ en) {
    int t = blockIdx.x * blockDim.x + threadIdx.x;
    if (t < EE) en[t] = dinv[ei[t]] * dinv[ei[EE + t]];
}

// h = log(x+1) @ exp_W + exp_b ; one block per node, 128 threads, 4 cols each
__global__ __launch_bounds__(128) void embed_k(const float* __restrict__ x,
                                               const float* __restrict__ W,
                                               const float* __restrict__ b,
                                               float* __restrict__ h) {
    __shared__ float lx[FF];
    int n = blockIdx.x;
    if (threadIdx.x < FF) lx[threadIdx.x] = logf(x[n * FF + threadIdx.x] + 1.0f);
    __syncthreads();
    int d = threadIdx.x * 4;
    F4 acc = *(const F4*)(b + d);
    #pragma unroll
    for (int f = 0; f < FF; f++) {
        float s = lx[f];
        F4 w = *(const F4*)(W + f * DV + d);
        acc.x += s * w.x; acc.y += s * w.y; acc.z += s * w.z; acc.w += s * w.w;
    }
    *(F4*)(h + (size_t)n * DV + d) = acc;
}

// ---------------- per-layer conversions to bf16 ----------------
// hb[n][d] = bf16(h[n][d]); rows >= NN zero-filled. 8 elems (16B out) per thread.
__global__ __launch_bounds__(256) void cvt_h_bf16(const float* __restrict__ h,
                                                  unsigned short* __restrict__ hb) {
    int t = blockIdx.x * blockDim.x + threadIdx.x;   // < NPAD*64
    int n = t >> 6, c8 = (t & 63) * 8;
    U4 o{0u, 0u, 0u, 0u};
    if (n < NN) {
        F4 v0 = *(const F4*)(h + (size_t)n * DV + c8);
        F4 v1 = *(const F4*)(h + (size_t)n * DV + c8 + 4);
        o = U4{pk2(v0.x, v0.y), pk2(v0.z, v0.w), pk2(v1.x, v1.y), pk2(v1.z, v1.w)};
    }
    *(U4*)(hb + (size_t)n * DV + c8) = o;
}

// wt[n][k] = bf16(W[k][n])  (transpose + convert, packed k-pairs)
__global__ __launch_bounds__(256) void cvt_w_bf16t(const float* __restrict__ W,
                                                   unsigned short* __restrict__ wt) {
    int t = blockIdx.x * blockDim.x + threadIdx.x;   // < DV*(DV/2)
    int n = t >> 8, kp = t & 255;
    float lo = W[(size_t)(2 * kp)     * DV + n];
    float hi = W[(size_t)(2 * kp + 1) * DV + n];
    ((unsigned*)wt)[(size_t)n * (DV / 2) + kp] = pk2(lo, hi);
}

// ---------------- WMMA GEMM: Y[N,512] = hb[N,512] @ wt^T  (bf16 in, f32 acc) ----------
constexpr int LDA = 40;   // shorts; 80B row stride -> conflict-free b128 frag loads

__global__ __launch_bounds__(256) void gemm_bf16(const unsigned short* __restrict__ hb,
                                                 const unsigned short* __restrict__ wt,
                                                 float* __restrict__ Y, int nRows) {
    __shared__ __align__(16) unsigned short As[BM * LDA];   // 128 x 32 bf16 (+pad)
    __shared__ __align__(16) unsigned short Bs[BN * LDA];   // 64  x 32 bf16 (+pad)

    const int tid   = threadIdx.x;
    const int lane  = tid & 31;
    const int wave  = tid >> 5;     // 0..7
    const int waveM = wave & 3;     // 4 waves over M (32 rows each)
    const int waveN = wave >> 2;    // 2 waves over N (32 cols each)
    const int m0 = blockIdx.x * BM;
    const int n0 = blockIdx.y * BN;

    const unsigned asBase = (unsigned)(size_t)As;
    const unsigned bsBase = (unsigned)(size_t)Bs;

    v8f acc[4];
    #pragma unroll
    for (int i = 0; i < 4; i++)
        #pragma unroll
        for (int j = 0; j < 8; j++) acc[i][j] = 0.0f;

    const int laneLo = lane & 15;
    const int k0a = (lane >> 4) * 8;    // A: k base in shorts (0 or 8)
    const int k0b = (lane >> 4) * 8;    // B: k base in shorts (0 or 8)

    for (int kk = 0; kk < DV; kk += BK) {
        // ---- async stage A tile: 128 rows x 64B (2 chunks/thread) ----
        #pragma unroll
        for (int i = 0; i < 2; i++) {
            int c   = i * 256 + tid;         // 0..511 chunks of 16B
            int row = c >> 2;
            int off = (c & 3) * 16;          // byte offset within row data
            const char* g = (const char*)hb + (((size_t)(m0 + row) * DV + kk) * 2 + off);
            asyncCp16(asBase + row * (LDA * 2) + off, g);
        }
        // ---- async stage B tile: 64 rows x 64B (1 chunk/thread) ----
        {
            int row = tid >> 2;
            int off = (tid & 3) * 16;
            const char* g = (const char*)wt + (((size_t)(n0 + row) * DV + kk) * 2 + off);
            asyncCp16(bsBase + row * (LDA * 2) + off, g);
        }
        waitAsync0();
        __syncthreads();

        // ---- fragments per ISA layout tables ----
        FragBF a[2], b[2];
        #pragma unroll
        for (int t = 0; t < 2; t++) {
            const unsigned short* p = &As[(waveM * 32 + t * 16 + laneLo) * LDA + k0a];
            a[t].f4[0] = *(const F4*)p;          // K k0..k0+7
            a[t].f4[1] = *(const F4*)(p + 16);   // K k0+16..k0+23
        }
        #pragma unroll
        for (int u = 0; u < 2; u++) {
            const unsigned short* p = &Bs[(waveN * 32 + u * 16 + laneLo) * LDA + k0b];
            b[u].f4[0] = *(const F4*)p;          // K k0..k0+7
            b[u].f4[1] = *(const F4*)(p + 16);   // K k0+16..k0+23
        }
        #pragma unroll
        for (int t = 0; t < 2; t++)
            #pragma unroll
            for (int u = 0; u < 2; u++)
                acc[t * 2 + u] = __builtin_amdgcn_wmma_f32_16x16x32_bf16(
                    false, a[t].v, false, b[u].v, (short)0, acc[t * 2 + u], false, false);
        __syncthreads();
    }

    // ---- epilogue: C/D layout (VGPR r -> row r | r+8) ----
    const int coln = lane & 15;
    const int rAdd = (lane >> 4) * 8;
    #pragma unroll
    for (int t = 0; t < 2; t++) {
        int tm = m0 + waveM * 32 + t * 16;
        #pragma unroll
        for (int u = 0; u < 2; u++) {
            int tn = n0 + waveN * 32 + u * 16 + coln;
            v8f c = acc[t * 2 + u];
            #pragma unroll
            for (int r = 0; r < 8; r++) {
                int row = tm + rAdd + r;
                if (row < nRows) Y[(size_t)row * DV + tn] = c[r];
            }
        }
    }
}

// ---------------- per-layer elementwise / edge kernels ----------------
__global__ __launch_bounds__(256) void init_agg(const float* __restrict__ xw,
                                                const float* __restrict__ selfn,
                                                const float* __restrict__ gb,
                                                float* __restrict__ agg) {
    int t = blockIdx.x * blockDim.x + threadIdx.x;
    int n = t >> 7, c = (t & 127) * 4;
    float s = selfn[n];
    F4 v  = *(const F4*)(xw + (size_t)n * DV + c);
    F4 bb = *(const F4*)(gb + c);
    F4 o{s * v.x + bb.x, s * v.y + bb.y, s * v.z + bb.z, s * v.w + bb.w};
    *(F4*)(agg + (size_t)n * DV + c) = o;
}

__global__ __launch_bounds__(256) void scatter_edges(const int* __restrict__ ei,
                                                     const float* __restrict__ en,
                                                     const float* __restrict__ xw,
                                                     float* __restrict__ agg) {
    int t = blockIdx.x * blockDim.x + threadIdx.x;
    int e = t >> 7, c = (t & 127) * 4;
    int src = ei[e];
    int dst = ei[EE + e];
    float w = en[e];
    F4 v = *(const F4*)(xw + (size_t)src * DV + c);
    float* p = agg + (size_t)dst * DV + c;
    atomAddF(p + 0, w * v.x);
    atomAddF(p + 1, w * v.y);
    atomAddF(p + 2, w * v.z);
    atomAddF(p + 3, w * v.w);
}

__global__ __launch_bounds__(256) void update_h(float* __restrict__ h,
                                                const float* __restrict__ agg) {
    int t = blockIdx.x * blockDim.x + threadIdx.x;
    size_t o = (size_t)(t >> 7) * DV + (t & 127) * 4;
    F4 hv = *(const F4*)(h + o);
    F4 av = *(const F4*)(agg + o);
    hv.x += fmaxf(av.x, 0.f); hv.y += fmaxf(av.y, 0.f);
    hv.z += fmaxf(av.z, 0.f); hv.w += fmaxf(av.w, 0.f);
    *(F4*)(h + o) = hv;
}

__global__ __launch_bounds__(256) void pool_k(const float* __restrict__ h,
                                              const int* __restrict__ batch,
                                              float* __restrict__ out) {
    int t = blockIdx.x * blockDim.x + threadIdx.x;
    int n = t >> 7, c = (t & 127) * 4;
    int g = batch[n];
    F4 v = *(const F4*)(h + (size_t)n * DV + c);
    float* p = out + (size_t)g * DV + c;
    atomAddF(p + 0, v.x);
    atomAddF(p + 1, v.y);
    atomAddF(p + 2, v.z);
    atomAddF(p + 3, v.w);
}

// ---------------- launcher ----------------
extern "C" void kernel_launch(void* const* d_in, const int* in_sizes, int n_in,
                              void* d_out, int out_size, void* d_ws, size_t ws_size,
                              hipStream_t stream) {
    const float* x     = (const float*)d_in[0];
    const int*   ei    = (const int*)  d_in[1];
    const int*   batch = (const int*)  d_in[2];
    const float* expW  = (const float*)d_in[3];
    const float* expB  = (const float*)d_in[4];
    const float* gcnW  = (const float*)d_in[5];
    const float* gcnB  = (const float*)d_in[6];
    float* out = (float*)d_out;

    // workspace layout (chunks packed back to back, no alignment padding)
    char* w = (char*)d_ws;
    const size_t ND   = (size_t)NN * DV * sizeof(float);          // f32 [N,512]
    const size_t NDH  = (size_t)NPAD * DV * sizeof(short);        // bf16 [NPAD,512]
    size_t off = 0;
    float*          h     = (float*)(w + off);          off += ND;
    float*          xw    = (float*)(w + off);          off += ND;
    float*          agg   = (float*)(w + off);          off += ND;
    unsigned short* hb    = (unsigned short*)(w + off); off += NDH;
    unsigned short* wtb   = (unsigned short*)(w + off); off += (size_t)DV * DV * sizeof(short);
    float*          dinv  = (float*)(w + off);          off += (size_t)NN * 4;
    float*          selfn = (float*)(w + off);          off += (size_t)NN * 4;
    float*          en    = (float*)(w + off);          off += (size_t)EE * 4;
    int*            degi  = (int*)(w + off);

    const int nodeBlocks = (NN * 128) / 256;    // 50000
    const int edgeBlocks = (EE * 128) / 256;    // 200000
    const int cvtHBlocks = (NPAD * 64) / 256;   // 25024

    // degree / norms
    zero_i32<<<(NN + 255) / 256, 256, 0, stream>>>(degi, NN);
    count_deg<<<(EE + 255) / 256, 256, 0, stream>>>(ei, degi);
    norms_k<<<(NN + 255) / 256, 256, 0, stream>>>(degi, dinv, selfn);
    enorm_k<<<(EE + 255) / 256, 256, 0, stream>>>(ei, dinv, en);

    // embedding
    embed_k<<<NN, 128, 0, stream>>>(x, expW, expB, h);

    // conv layers
    dim3 ggrid(NPAD / BM, DV / BN);
    for (int l = 0; l < LL; l++) {
        cvt_h_bf16<<<cvtHBlocks, 256, 0, stream>>>(h, hb);
        cvt_w_bf16t<<<(DV * DV / 2) / 256, 256, 0, stream>>>(gcnW + (size_t)l * DV * DV, wtb);
        gemm_bf16<<<ggrid, 256, 0, stream>>>(hb, wtb, xw, NN);
        init_agg<<<nodeBlocks, 256, 0, stream>>>(xw, selfn, gcnB + (size_t)l * DV, agg);
        scatter_edges<<<edgeBlocks, 256, 0, stream>>>(ei, en, xw, agg);
        update_h<<<nodeBlocks, 256, 0, stream>>>(h, agg);
    }

    // global add pool
    zero_f32<<<(GG * DV + 255) / 256, 256, 0, stream>>>(out, GG * DV);
    pool_k<<<nodeBlocks, 256, 0, stream>>>(h, batch, out);
}
